// MultiheadAttention_34385508172202
// MI455X (gfx1250) — matrix operand order repaired and checked by reading, not compiled, not run on hardware
//
#include <hip/hip_runtime.h>

// ---------------- problem constants ----------------
#define DMODEL 1024
#define NHEAD  16
#define DK     64
#define SEQ    2048
#define BATCH  4
#define MTOT   (BATCH * SEQ)        // 8192 rows
#define BH     (BATCH * NHEAD)      // 64 batch-heads

typedef __attribute__((ext_vector_type(16))) _Float16 v16h;
typedef __attribute__((ext_vector_type(8)))  _Float16 v8h;
typedef __attribute__((ext_vector_type(8)))  float    v8f;
typedef __attribute__((ext_vector_type(4)))  float    v4f;
typedef __attribute__((ext_vector_type(4)))  int      v4i;

// D = A(16x32 f16) * B(32x16 f16) + C(16x16 f32)
__device__ __forceinline__ v8f wmma32(v16h a, v16h b, v8f c) {
    return __builtin_amdgcn_wmma_f32_16x16x32_f16(
        /*neg_a=*/false, a, /*neg_b=*/false, b,
        /*c_mod=*/(short)0, c, /*reuse_a=*/false, /*reuse_b=*/false);
}

__device__ __forceinline__ v16h cat8(v8h lo, v8h hi) {
    v16h r;
#pragma unroll
    for (int i = 0; i < 8; ++i) { r[i] = lo[i]; r[i + 8] = hi[i]; }
    return r;
}

// A-fragment (16x32) from fp32 row-major X. `p` -> (row m, k = k0 + g*8).
// Halves 0..7 <- p[0..7], halves 8..15 <- p[16..23].
__device__ __forceinline__ v16h load_a_f32(const float* __restrict__ p) {
    v4f f0 = *(const v4f*)(p);
    v4f f1 = *(const v4f*)(p + 4);
    v4f f2 = *(const v4f*)(p + 16);
    v4f f3 = *(const v4f*)(p + 20);
    v16h a;
#pragma unroll
    for (int i = 0; i < 4; ++i) {
        a[i]      = (_Float16)f0[i];
        a[i + 4]  = (_Float16)f1[i];
        a[i + 8]  = (_Float16)f2[i];
        a[i + 12] = (_Float16)f3[i];
    }
    return a;
}

// ---------------- kernel 1: fp32 -> f16 conversion (b128 in/out) -----------
__global__ void cvt_f32_f16(const float* __restrict__ src,
                            _Float16* __restrict__ dst, int n8) {
    int i = blockIdx.x * blockDim.x + threadIdx.x;
    const int stride = gridDim.x * blockDim.x;
    for (; i < n8; i += stride) {
        const float* s = src + 8 * (size_t)i;
        v4f f0 = *(const v4f*)(s);
        v4f f1 = *(const v4f*)(s + 4);
        v8h h;
#pragma unroll
        for (int j = 0; j < 4; ++j) {
            h[j]     = (_Float16)f0[j];
            h[j + 4] = (_Float16)f1[j];
        }
        *(v8h*)(dst + 8 * (size_t)i) = h;
    }
}

// ---------------- kernel 2: projection GEMM (double-buffered) --------------
// C[m,n] = sum_k X[m,k] * W[n,k] + bias[n]; X fp32, W f16.
// Wave tile: 32x64 (2 M-subtiles x 4 N-tiles), 8 WMMAs per 32-wide k-step.
// transposeV==0: store f16 [bh][s][64] ; ==1: store f16 [bh][64][s].
__global__ __launch_bounds__(128) void proj_gemm(
    const float* __restrict__ X, const _Float16* __restrict__ W,
    const float* __restrict__ bias, _Float16* __restrict__ Out, int transposeV) {
    const int lane = threadIdx.x & 31;
    const int wave = threadIdx.x >> 5;
    const int g = lane >> 4, r = lane & 15;
    const int m0 = (blockIdx.y * 4 + wave) * 32;
    const int n0 = blockIdx.x * 64;

    const float* ap0 = X + (size_t)(m0 + r) * DMODEL + g * 8;
    const float* ap1 = ap0 + (size_t)16 * DMODEL;
    const _Float16* bp = W + (size_t)(n0 + r) * DMODEL + g * 8;

    v8f acc[2][4] = {};
    v16h a[2], b[4];
    a[0] = load_a_f32(ap0);
    a[1] = load_a_f32(ap1);
#pragma unroll
    for (int nb = 0; nb < 4; ++nb) b[nb] = cat8(*(const v8h*)(bp + (size_t)nb * 16 * DMODEL), *(const v8h*)(bp + (size_t)nb * 16 * DMODEL + 16));

    for (int k0 = 0; k0 < DMODEL; k0 += 32) {
        const int kn = (k0 + 32 < DMODEL) ? (k0 + 32) : 0;  // harmless reload on last iter
        v16h an[2], bn[4];
        an[0] = load_a_f32(ap0 + kn);
        an[1] = load_a_f32(ap1 + kn);
#pragma unroll
        for (int nb = 0; nb < 4; ++nb)
            bn[nb] = cat8(*(const v8h*)(bp + (size_t)nb * 16 * DMODEL + kn), *(const v8h*)(bp + (size_t)nb * 16 * DMODEL + kn + 16));
#pragma unroll
        for (int nb = 0; nb < 4; ++nb)
#pragma unroll
            for (int mi = 0; mi < 2; ++mi)
                acc[mi][nb] = wmma32(a[mi], b[nb], acc[mi][nb]);
        a[0] = an[0]; a[1] = an[1];
#pragma unroll
        for (int nb = 0; nb < 4; ++nb) b[nb] = bn[nb];
    }

#pragma unroll
    for (int mi = 0; mi < 2; ++mi) {
        const int mrow0 = m0 + mi * 16 + g * 8;     // 8 consecutive rows per lane
        const int bidx  = mrow0 >> 11;              // batch index
        const int s0    = mrow0 & (SEQ - 1);
#pragma unroll
        for (int nb = 0; nb < 4; ++nb) {
            const int n = n0 + nb * 16 + r;
            const float bv = bias[n];
            const int h = n >> 6, d = n & 63;
            if (!transposeV) {
                _Float16* op = Out + (((size_t)(bidx * NHEAD + h) * SEQ + s0) * DK + d);
#pragma unroll
                for (int rr = 0; rr < 8; ++rr)
                    op[(size_t)rr * DK] = (_Float16)(acc[mi][nb][rr] + bv);
            } else {
                v8h pk;
#pragma unroll
                for (int rr = 0; rr < 8; ++rr) pk[rr] = (_Float16)(acc[mi][nb][rr] + bv);
                *(v8h*)(Out + (((size_t)(bidx * NHEAD + h) * DK + d) * SEQ + s0)) = pk;
            }
        }
    }
}

// ---------------- kernel 3: flash attention (register resident) ------------
// Qp,Kp: [bh][s][64] f16 ; Vt: [bh][64][s] f16 ; mask: [S][S] int ; Ctx: [m][1024] f16
// S^T = K*Q^T so exp(S) lands directly in the P*V A-fragment layout.
// K-fragments are double-buffered across s_k chunks.
__global__ __launch_bounds__(128) void attn_kernel(
    const _Float16* __restrict__ Qp, const _Float16* __restrict__ Kp,
    const _Float16* __restrict__ Vt, const int* __restrict__ mask,
    _Float16* __restrict__ Ctx) {
    const int lane = threadIdx.x & 31;
    const int wave = threadIdx.x >> 5;
    const int g = lane >> 4, r = lane & 15;
    const int bh = blockIdx.y;
    const int q0 = (blockIdx.x * 4 + wave) * 16;

    const _Float16* Qb = Qp + (size_t)bh * SEQ * DK;
    const _Float16* Kb = Kp + (size_t)bh * SEQ * DK;
    const _Float16* Vb = Vt + (size_t)bh * DK * SEQ;

    // Q as B-fragments (loop invariant): lane -> q column q0+r, d = c*32+{g*8, 16+g*8}
    v16h qf[2];
#pragma unroll
    for (int c = 0; c < 2; ++c)
        qf[c] = cat8(*(const v8h*)(Qb + (size_t)(q0 + r) * DK + c * 32 + g * 8), *(const v8h*)(Qb + (size_t)(q0 + r) * DK + c * 32 + g * 8 + 16));

    float mrun = -3.0e38f, lrun = 0.0f;
    v8f acc[4] = {};

    // Preload K A-fragments for chunk 0: kc[t][c], rows s_k = t*16+r, k = c*32+{g*8, 16+g*8}
    v16h kc[2][2];
#pragma unroll
    for (int t = 0; t < 2; ++t) {
        const _Float16* kp = Kb + (size_t)(t * 16 + r) * DK + g * 8;
        kc[t][0] = cat8(*(const v8h*)(kp),      *(const v8h*)(kp + 16));
        kc[t][1] = cat8(*(const v8h*)(kp + 32), *(const v8h*)(kp + 48));
    }

    for (int j0 = 0; j0 < SEQ; j0 += 32) {
        const int jn = (j0 + 32 < SEQ) ? (j0 + 32) : 0;  // harmless reload on last iter

        // Issue next chunk's K loads and this chunk's V + mask loads up front.
        v16h kn_[2][2];
#pragma unroll
        for (int t = 0; t < 2; ++t) {
            const _Float16* kp = Kb + (size_t)(jn + t * 16 + r) * DK + g * 8;
            kn_[t][0] = cat8(*(const v8h*)(kp),      *(const v8h*)(kp + 16));
            kn_[t][1] = cat8(*(const v8h*)(kp + 32), *(const v8h*)(kp + 48));
        }
        v16h vb[4];
#pragma unroll
        for (int t4 = 0; t4 < 4; ++t4)
            vb[t4] = cat8(*(const v8h*)(Vb + (size_t)(t4 * 16 + r) * SEQ + j0 + g * 8), *(const v8h*)(Vb + (size_t)(t4 * 16 + r) * SEQ + j0 + g * 8 + 16));
        v4i mA[2], mB[2];
#pragma unroll
        for (int t = 0; t < 2; ++t) {
            const int* mp = mask + (size_t)(q0 + r) * SEQ + j0 + t * 16 + g * 8;
            mA[t] = *(const v4i*)(mp);
            mB[t] = *(const v4i*)(mp + 4);
        }

        // S^T tiles (rows = s_k, cols = q) from the current K buffer.
        v8f st[2];
#pragma unroll
        for (int t = 0; t < 2; ++t) {
            v8f z = {};
            st[t] = wmma32(kc[t][0], qf[0], z);
            st[t] = wmma32(kc[t][1], qf[1], st[t]);
        }

        // scale + mask; lane holds q = q0+r, s_k = j0 + t*16 + g*8 + rr
        float pv[2][8];
        float cmax = -3.0e38f;
#pragma unroll
        for (int t = 0; t < 2; ++t)
#pragma unroll
            for (int rr = 0; rr < 8; ++rr) {
                const int mv = (rr < 4) ? mA[t][rr] : mB[t][rr - 4];
                float sv = st[t][rr] * 0.125f;            // 1/sqrt(64)
                sv = (mv == 0) ? -1.0e9f : sv;
                pv[t][rr] = sv;
                cmax = fmaxf(cmax, sv);
            }
        cmax = fmaxf(cmax, __shfl_xor(cmax, 16, 32));
        const float mnew = fmaxf(mrun, cmax);
        const float corr = __expf(mrun - mnew);

        float psum = 0.0f;
        v16h pf;
#pragma unroll
        for (int t = 0; t < 2; ++t)
#pragma unroll
            for (int rr = 0; rr < 8; ++rr) {
                const float e = __expf(pv[t][rr] - mnew);
                psum += e;
                pf[t * 8 + rr] = (_Float16)e;   // exactly the A-fragment layout
            }
        psum += __shfl_xor(psum, 16, 32);
        lrun = lrun * corr + psum;
        mrun = mnew;

        // rescale O accumulators (O rows are q = g*8+rr -> broadcast corr)
#pragma unroll
        for (int rr = 0; rr < 8; ++rr) {
            const float co = __shfl(corr, g * 8 + rr, 32);
#pragma unroll
            for (int t4 = 0; t4 < 4; ++t4) acc[t4][rr] *= co;
        }

        // O += P * V   (B-fragments contiguous thanks to transposed V)
#pragma unroll
        for (int t4 = 0; t4 < 4; ++t4)
            acc[t4] = wmma32(pf, vb[t4], acc[t4]);

        // rotate K double-buffer
#pragma unroll
        for (int t = 0; t < 2; ++t) { kc[t][0] = kn_[t][0]; kc[t][1] = kn_[t][1]; }
    }

    const float linv = 1.0f / lrun;
    const int b = bh >> 4, h = bh & (NHEAD - 1);
    const size_t row0 = (size_t)b * SEQ + q0 + g * 8;
#pragma unroll
    for (int rr = 0; rr < 8; ++rr) {
        const float li = __shfl(linv, g * 8 + rr, 32);
#pragma unroll
        for (int t4 = 0; t4 < 4; ++t4)
            Ctx[(row0 + rr) * DMODEL + h * DK + t4 * 16 + r] =
                (_Float16)(acc[t4][rr] * li);
    }
}

// ---------------- kernel 4: output GEMM (f16 A, fp32 out, double-buffered) -
__global__ __launch_bounds__(128) void out_gemm(
    const _Float16* __restrict__ A, const _Float16* __restrict__ W,
    const float* __restrict__ bias, float* __restrict__ Out) {
    const int lane = threadIdx.x & 31;
    const int wave = threadIdx.x >> 5;
    const int g = lane >> 4, r = lane & 15;
    const int m0 = (blockIdx.y * 4 + wave) * 32;
    const int n0 = blockIdx.x * 64;

    const _Float16* ap0 = A + (size_t)(m0 + r) * DMODEL + g * 8;
    const _Float16* ap1 = ap0 + (size_t)16 * DMODEL;
    const _Float16* bp  = W + (size_t)(n0 + r) * DMODEL + g * 8;

    v8f acc[2][4] = {};
    v16h a[2], b[4];
    a[0] = cat8(*(const v8h*)(ap0), *(const v8h*)(ap0 + 16));
    a[1] = cat8(*(const v8h*)(ap1), *(const v8h*)(ap1 + 16));
#pragma unroll
    for (int nb = 0; nb < 4; ++nb) b[nb] = cat8(*(const v8h*)(bp + (size_t)nb * 16 * DMODEL), *(const v8h*)(bp + (size_t)nb * 16 * DMODEL + 16));

    for (int k0 = 0; k0 < DMODEL; k0 += 32) {
        const int kn = (k0 + 32 < DMODEL) ? (k0 + 32) : 0;
        v16h an[2], bn[4];
        an[0] = cat8(*(const v8h*)(ap0 + kn), *(const v8h*)(ap0 + kn + 16));
        an[1] = cat8(*(const v8h*)(ap1 + kn), *(const v8h*)(ap1 + kn + 16));
#pragma unroll
        for (int nb = 0; nb < 4; ++nb)
            bn[nb] = cat8(*(const v8h*)(bp + (size_t)nb * 16 * DMODEL + kn), *(const v8h*)(bp + (size_t)nb * 16 * DMODEL + kn + 16));
#pragma unroll
        for (int nb = 0; nb < 4; ++nb)
#pragma unroll
            for (int mi = 0; mi < 2; ++mi)
                acc[mi][nb] = wmma32(a[mi], b[nb], acc[mi][nb]);
        a[0] = an[0]; a[1] = an[1];
#pragma unroll
        for (int nb = 0; nb < 4; ++nb) b[nb] = bn[nb];
    }

#pragma unroll
    for (int mi = 0; mi < 2; ++mi) {
        const int mrow0 = m0 + mi * 16 + g * 8;
#pragma unroll
        for (int nb = 0; nb < 4; ++nb) {
            const int n = n0 + nb * 16 + r;
            const float bv = bias[n];
            float* op = Out + (size_t)mrow0 * DMODEL + n;
#pragma unroll
            for (int rr = 0; rr < 8; ++rr)
                op[(size_t)rr * DMODEL] = acc[mi][nb][rr] + bv;
        }
    }
}

// ---------------- host-side launcher ----------------
extern "C" void kernel_launch(void* const* d_in, const int* in_sizes, int n_in,
                              void* d_out, int out_size, void* d_ws, size_t ws_size,
                              hipStream_t stream) {
    const float* q    = (const float*)d_in[0];
    const float* k    = (const float*)d_in[1];
    const float* v    = (const float*)d_in[2];
    const int*   mask = (const int*)d_in[3];
    const float* wQ   = (const float*)d_in[4];
    const float* bQ   = (const float*)d_in[5];
    const float* wK   = (const float*)d_in[6];
    const float* bK   = (const float*)d_in[7];
    const float* wV   = (const float*)d_in[8];
    const float* bV   = (const float*)d_in[9];
    const float* wO   = (const float*)d_in[10];
    const float* bO   = (const float*)d_in[11];
    (void)in_sizes; (void)n_in; (void)out_size; (void)ws_size;

    char* ws = (char*)d_ws;
    size_t off = 0;
    auto carve = [&](size_t bytes) -> char* {
        char* p = ws + off;
        off += (bytes + 255) & ~(size_t)255;
        return p;
    };
    const size_t WB = (size_t)DMODEL * DMODEL * sizeof(_Float16);   // 2 MB
    const size_t PB = (size_t)BH * SEQ * DK * sizeof(_Float16);     // 16 MB
    const size_t CB = (size_t)MTOT * DMODEL * sizeof(_Float16);     // 16 MB
    _Float16* wq16 = (_Float16*)carve(WB);
    _Float16* wk16 = (_Float16*)carve(WB);
    _Float16* wv16 = (_Float16*)carve(WB);
    _Float16* wo16 = (_Float16*)carve(WB);
    _Float16* Qp   = (_Float16*)carve(PB);
    _Float16* Kp   = (_Float16*)carve(PB);
    _Float16* Vt   = (_Float16*)carve(PB);
    _Float16* ctx  = (_Float16*)carve(CB);

    const int n8 = (DMODEL * DMODEL) / 8;   // 131072
    cvt_f32_f16<<<512, 256, 0, stream>>>(wQ, wq16, n8);
    cvt_f32_f16<<<512, 256, 0, stream>>>(wK, wk16, n8);
    cvt_f32_f16<<<512, 256, 0, stream>>>(wV, wv16, n8);
    cvt_f32_f16<<<512, 256, 0, stream>>>(wO, wo16, n8);

    dim3 blk(128);
    dim3 pg(DMODEL / 64, MTOT / 128);  // (16, 64): 64-col x 128-row block tiles
    proj_gemm<<<pg, blk, 0, stream>>>(q, wq16, bQ, Qp, 0);
    proj_gemm<<<pg, blk, 0, stream>>>(k, wk16, bK, Kp, 0);
    proj_gemm<<<pg, blk, 0, stream>>>(v, wv16, bV, Vt, 1);

    attn_kernel<<<dim3(SEQ / 64, BH), blk, 0, stream>>>(Qp, Kp, Vt, mask, ctx);

    out_gemm<<<pg, blk, 0, stream>>>(ctx, wo16, bO, (float*)d_out);
}
